// ManualLSTMCell_32598801776688
// MI455X (gfx1250) — matrix operand order repaired and checked by reading, not compiled, not run on hardware
//
#include <hip/hip_runtime.h>

// ---------------------------------------------------------------------------
// Fused LSTM cell for MI455X (gfx1250, wave32).
//   B=4096, IN=HS=1024.  68.7 GFLOP of GEMM, ~112MB HBM footprint.
//   Compute-bound at bf16 WMMA rates -> convert once to bf16 (32MB working
//   set is fully L2-resident, 192MB L2), then one fused kernel:
//     - per wave: 2 M-tiles x 1 N-tile x 4 gates = 8 fp32 accumulators
//     - weights staged in LDS via global_load_async_to_lds_b128 (ASYNCcnt),
//       double-buffered 128-K chunks, ds_load_b128 fragment reads
//     - elementwise LSTM epilogue fused on the C/D fragment layout
// ---------------------------------------------------------------------------

#define B_DIM   4096
#define IN_DIM  1024
#define HS_DIM  1024

#define XN_ELEMS (4096 * 1024)          // elements in x (== h)
#define WN_ELEMS (1024 * 1024)          // elements per weight matrix

#define KCHUNK    128                    // K elements staged per chunk
#define KPAD      136                    // row stride in LDS (+16B pad, bank-conflict-free)
#define BUF_ELEMS (64 * KPAD)            // 4 gates x 16 rows, one chunk: 8704 elems
#define NCHUNK    16                     // 2048 total K / 128

typedef __attribute__((ext_vector_type(16))) __bf16       v16bf;
typedef __attribute__((ext_vector_type(8)))  float        v8f;
typedef __attribute__((ext_vector_type(4)))  unsigned int v4u;

union FragU {
    v4u   u[2];
    v16bf v;
};

// A fragment (16x32 bf16, MxK) from row-major memory, two 16B chunks at +0/+16
// elems (lanes 0-15: K 0..7 & 16..23; lanes 16-31 pre-offset +8 -> 8..15 & 24..31).
__device__ __forceinline__ v16bf load_frag_a(const unsigned short* p) {
    FragU f;
    f.u[0] = *reinterpret_cast<const v4u*>(p);
    f.u[1] = *reinterpret_cast<const v4u*>(p + 16);
    return f.v;
}

// B fragment (32x16 bf16, KxN): lane n holds W[n, k0..k0+7] & W[n, k0+16..k0+23]
// (lanes 16-31 pre-offset +8 in K by the caller). Source is LDS.
__device__ __forceinline__ v16bf load_frag_b(const unsigned short* p) {
    FragU f;
    f.u[0] = *reinterpret_cast<const v4u*>(p);
    f.u[1] = *reinterpret_cast<const v4u*>(p + 16);
    return f.v;
}

__device__ __forceinline__ float sigmoidf_fast(float x) {
    return 1.0f / (1.0f + __expf(-x));
}

// ---------------------------------------------------------------------------
// fp32 -> bf16 (round to nearest even), 4 elements per thread.
// ---------------------------------------------------------------------------
__device__ __forceinline__ unsigned int f2bf_bits(unsigned int u) {
    return (u + 0x7FFFu + ((u >> 16) & 1u)) >> 16;
}

__global__ __launch_bounds__(256) void f32_to_bf16_kernel(
    const float* __restrict__ src, unsigned short* __restrict__ dst, int n) {
    int i = (blockIdx.x * blockDim.x + threadIdx.x) * 4;
    if (i >= n) return;
    const uint4 u = *reinterpret_cast<const uint4*>(src + i);
    uint2 o;
    o.x = f2bf_bits(u.x) | (f2bf_bits(u.y) << 16);
    o.y = f2bf_bits(u.z) | (f2bf_bits(u.w) << 16);
    *reinterpret_cast<uint2*>(dst + i) = o;
}

// ---------------------------------------------------------------------------
// Fused LSTM GEMM kernel.
//   WG = 256 threads = 8 waves; wave w owns rows [blockIdx.y*256 + w*32, +32)
//   (two 16-row M-tiles), all waves share the 16-column N-tile blockIdx.x.
//   Weight slice for the N-tile is staged in LDS (async DMA, double-buffered).
// ---------------------------------------------------------------------------
__global__ __launch_bounds__(256) void lstm_wmma_kernel(
    const unsigned short* __restrict__ xh,    // [2][4096][1024] bf16: x then h
    const unsigned short* __restrict__ wall,  // [8][1024][1024] bf16: ix,fx,gx,ox,ih,fh,gh,oh
    const float* __restrict__ bix, const float* __restrict__ bfx,
    const float* __restrict__ bgx, const float* __restrict__ box,
    const float* __restrict__ bih, const float* __restrict__ bfh,
    const float* __restrict__ bgh, const float* __restrict__ boh,
    const float* __restrict__ c_in,
    float* __restrict__ h_out, float* __restrict__ c_out) {

    __shared__ __align__(16) unsigned short smem[2 * BUF_ELEMS];

    const int tid  = threadIdx.x;
    const int wave = tid >> 5;
    const int lane = tid & 31;
    const int half = lane >> 4;   // 0: lanes 0-15, 1: lanes 16-31
    const int l15  = lane & 15;

    const int n0 = blockIdx.x << 4;                  // output column tile
    const int m0 = blockIdx.y * 256 + wave * 32;     // first of two 16-row tiles

    // LDS byte address of smem for the async-DMA destination operand.
    const unsigned lds_base =
        (unsigned)(unsigned long long)(__attribute__((address_space(3))) unsigned short*)smem;

    // ---- async fill of one 128-K weight chunk into one LDS buffer ----------
    // chunk c: phase = c>>3 (0 = x-weights, 1 = h-weights), kbase = (c&7)*128.
    // 64 rows (gate*16 + r) x 256B, each thread DMAs 4 x 16B units.
    auto fill_chunk = [&](int c) {
        const int phase = c >> 3;
        const int kbase = (c & 7) * KCHUNK;
        const unsigned bufb = lds_base + (unsigned)((c & 1) * BUF_ELEMS * 2);
        const size_t wphase = (size_t)(phase * 4) * (size_t)WN_ELEMS;
#pragma unroll
        for (int j = 0; j < 4; ++j) {
            const int u    = tid + 256 * j;
            const int row  = u >> 4;            // gate*16 + r
            const int kb   = (u & 15) * 8;      // elems within 128-K row
            const int nidx = ((row >> 4) << 10) + n0 + (row & 15);
            const unsigned long long gaddr = (unsigned long long)
                (wall + wphase + (size_t)nidx * 1024 + (size_t)(kbase + kb));
            const unsigned laddr = bufb + (unsigned)(row * (KPAD * 2) + kb * 2);
            asm volatile("global_load_async_to_lds_b128 %0, %1, off"
                         :: "v"(laddr), "v"(gaddr) : "memory");
        }
    };

    v8f acc[4][2];
#pragma unroll
    for (int g = 0; g < 4; ++g) {
        acc[g][0] = {};
        acc[g][1] = {};
    }

    // Per-lane A base offset (fragment layout; see load_frag_a).
    const size_t a_off0 = (size_t)(m0 + l15) * 1024 + (size_t)(half * 8);
    // Per-lane LDS B fragment base (fragment layout; see load_frag_b).
    const unsigned short* sb_lane = smem + l15 * KPAD + half * 8;

    fill_chunk(0);

    for (int c = 0; c < NCHUNK; ++c) {
        asm volatile("s_wait_asynccnt 0x0" ::: "memory");
        __syncthreads();
        if (c + 1 < NCHUNK) fill_chunk(c + 1);

        const int phase = c >> 3;
        const int kbase = (c & 7) * KCHUNK;
        const unsigned short* pa = xh + (size_t)phase * XN_ELEMS + a_off0 + kbase;
        const unsigned short* sb = sb_lane + (c & 1) * BUF_ELEMS;

#pragma unroll
        for (int ks = 0; ks < KCHUNK; ks += 32) {
            v16bf a0 = load_frag_a(pa + ks);
            v16bf a1 = load_frag_a(pa + 16 * 1024 + ks);
#pragma unroll
            for (int g = 0; g < 4; ++g) {
                v16bf b = load_frag_b(sb + g * (16 * KPAD) + ks);
                acc[g][0] = __builtin_amdgcn_wmma_f32_16x16x32_bf16(
                    false, a0, false, b, (short)0, acc[g][0], false, false);
                acc[g][1] = __builtin_amdgcn_wmma_f32_16x16x32_bf16(
                    false, a1, false, b, (short)0, acc[g][1], false, false);
            }
        }
    }

    // ---- elementwise LSTM epilogue on the C/D fragment layout --------------
    // (m, n): lanes 0-15 -> VGPR r holds M=r; lanes 16-31 -> M=8+r. N = l15.
    const int n = n0 + l15;
    const float cbi = bix[n] + bih[n];
    const float cbf = bfx[n] + bfh[n];
    const float cbg = bgx[n] + bgh[n];
    const float cbo = box[n] + boh[n];

#pragma unroll
    for (int t = 0; t < 2; ++t) {
        const int mbase = m0 + t * 16 + half * 8;
#pragma unroll
        for (int r = 0; r < 8; ++r) {
            const size_t idx = (size_t)(mbase + r) * HS_DIM + n;
            const float iv = sigmoidf_fast(acc[0][t][r] + cbi);
            const float fv = sigmoidf_fast(acc[1][t][r] + cbf);
            const float gv = tanhf(acc[2][t][r] + cbg);
            const float ov = sigmoidf_fast(acc[3][t][r] + cbo);
            const float cn = fv * c_in[idx] + iv * gv;
            const float hn = ov * tanhf(cn);
            c_out[idx] = cn;
            h_out[idx] = hn;
        }
    }
}

// ---------------------------------------------------------------------------
// Launcher
// ---------------------------------------------------------------------------
extern "C" void kernel_launch(void* const* d_in, const int* in_sizes, int n_in,
                              void* d_out, int out_size, void* d_ws, size_t ws_size,
                              hipStream_t stream) {
    (void)in_sizes; (void)n_in; (void)out_size; (void)ws_size;

    // d_in order: 0 x, 1 h, 2 c, 3 Wix, 4 bix, 5 Wfx, 6 bfx, 7 Wgx, 8 bgx,
    //             9 Wox, 10 box, 11 Wih, 12 bih, 13 Wfh, 14 bfh, 15 Wgh,
    //             16 bgh, 17 Woh, 18 boh
    unsigned short* ws = (unsigned short*)d_ws;
    unsigned short* xh = ws;                        // [2*XN]: x then h (contiguous)
    unsigned short* wb = xh + 2 * (size_t)XN_ELEMS; // [8*WN]: ix,fx,gx,ox,ih,fh,gh,oh

    const int wsrc_idx[8] = {3, 5, 7, 9, 11, 13, 15, 17};

    // --- convert fp32 -> bf16 ---
    {
        const int threads = 256;
        const int gx = (XN_ELEMS / 4 + threads - 1) / threads;
        f32_to_bf16_kernel<<<gx, threads, 0, stream>>>((const float*)d_in[0], xh, XN_ELEMS);
        f32_to_bf16_kernel<<<gx, threads, 0, stream>>>((const float*)d_in[1],
                                                       xh + (size_t)XN_ELEMS, XN_ELEMS);
        const int gw = (WN_ELEMS / 4 + threads - 1) / threads;
        for (int w = 0; w < 8; ++w) {
            f32_to_bf16_kernel<<<gw, threads, 0, stream>>>(
                (const float*)d_in[wsrc_idx[w]], wb + (size_t)w * WN_ELEMS, WN_ELEMS);
        }
    }

    // --- fused LSTM GEMM + epilogue ---
    float* h_out = (float*)d_out;
    float* c_out = h_out + (size_t)B_DIM * HS_DIM;

    dim3 grid(HS_DIM / 16, B_DIM / 256);  // (64, 16)
    dim3 block(256);
    lstm_wmma_kernel<<<grid, block, 0, stream>>>(
        xh, wb,
        (const float*)d_in[4],  (const float*)d_in[6],
        (const float*)d_in[8],  (const float*)d_in[10],
        (const float*)d_in[12], (const float*)d_in[14],
        (const float*)d_in[16], (const float*)d_in[18],
        (const float*)d_in[2],
        h_out, c_out);
}
